// Head_74655121539258
// MI455X (gfx1250) — matrix operand order repaired and checked by reading, not compiled, not run on hardware
//
#include <hip/hip_runtime.h>
#include <stdint.h>

// Problem constants (match reference)
#define EMB   384
#define HDIM  64
#define BB    512
#define TT    256
#define N3    192           // Q|K|V concatenated out-features
#define NKC   12            // 384 / 32 K-chunks

typedef __attribute__((ext_vector_type(16))) __bf16 v16bf;
typedef __attribute__((ext_vector_type(8)))  float  v8f;
typedef unsigned short ushort_t;

union Frag16 {
    v16bf v;
    ushort_t s[16];
    uint4 q2[2];
};

static __device__ __forceinline__ ushort_t f2bf(float f) {
    union { float f; unsigned u; } x; x.f = f;
    unsigned r = x.u + 0x7FFFu + ((x.u >> 16) & 1u);   // round-to-nearest-even
    return (ushort_t)(r >> 16);
}

// Low 32 bits of a flat LDS pointer == LDS byte offset (aperture keeps addr[31:0])
static __device__ __forceinline__ unsigned lds_off(const void* p) {
    return (unsigned)(uintptr_t)p;
}

// CDNA5 async global->LDS copy, 16 bytes per lane (ASYNCcnt-tracked)
static __device__ __forceinline__ void async_b128(unsigned lds, const void* g) {
    asm volatile("global_load_async_to_lds_b128 %0, %1, off"
                 :: "v"(lds), "v"(g) : "memory");
}

// -----------------------------------------------------------------------------
// Kernel 1: fused QKV projection.  Y = x @ [WQ;WK;WV]^T  (M=B*T, N=192, K=384)
// Outputs bf16: Q [B,T,64], K [B,T,64], V transposed [B,64,T].
// Block = 256 threads (8 waves); block covers 128 rows; wave w owns rows
// [w*16, w*16+16) and all 12 N-tiles.
// -----------------------------------------------------------------------------
__global__ __launch_bounds__(256, 1)
void qkv_proj_kernel(const float* __restrict__ x,
                     const float* __restrict__ wq,
                     const float* __restrict__ wk,
                     const float* __restrict__ wv,
                     ushort_t* __restrict__ qb,
                     ushort_t* __restrict__ kb,
                     ushort_t* __restrict__ vt) {
    __shared__ ushort_t xs[128 * 32];   // x chunk, bf16 (8 KB)
    __shared__ ushort_t wl[N3 * 32];    // weight chunk, bf16 (12 KB)

    const int tid  = threadIdx.x;
    const int lane = tid & 31;
    const int w    = tid >> 5;
    const int mbase = blockIdx.x * 128;

    v8f acc[12];
#pragma unroll
    for (int nt = 0; nt < 12; ++nt)
#pragma unroll
        for (int i = 0; i < 8; ++i) acc[nt][i] = 0.0f;

    for (int kc = 0; kc < NKC; ++kc) {
        __syncthreads();
#pragma unroll
        for (int j = 0; j < 16; ++j) {
            int e   = tid + 256 * j;             // 0..4095
            int row = e >> 5, col = e & 31;
            xs[row * 32 + col] =
                f2bf(x[(size_t)(mbase + row) * EMB + kc * 32 + col]);
        }
#pragma unroll
        for (int j = 0; j < 24; ++j) {
            int e = tid + 256 * j;               // 0..6143
            int n = e >> 5, col = e & 31;
            const float* wp = (n < 64) ? wq : (n < 128) ? wk : wv;
            wl[n * 32 + col] = f2bf(wp[(n & 63) * EMB + kc * 32 + col]);
        }
        __syncthreads();

        // A fragment: 16x32 bf16, row = lane&15, halves: K = h + 16*(lane>>4)
        Frag16 a;
        {
            const int row = w * 16 + (lane & 15);
            const int c0  = (lane < 16) ? 0 : 16;
            const uint4* p = (const uint4*)&xs[row * 32];
            a.q2[0] = p[(c0 >> 3)];
            a.q2[1] = p[(c0 >> 3) + 1];
        }
#pragma unroll
        for (int nt = 0; nt < 12; ++nt) {
            // B fragment: 32x16 bf16, col = lane&15, halves: K = h + 16*(lane>>4)
            Frag16 bf;
            const int n  = nt * 16 + (lane & 15);
            const int c0 = (lane < 16) ? 0 : 16;
            const uint4* p = (const uint4*)&wl[n * 32 + c0];
            bf.q2[0] = p[0];
            bf.q2[1] = p[1];
            acc[nt] = __builtin_amdgcn_wmma_f32_16x16x32_bf16(
                false, a.v, false, bf.v, (short)0, acc[nt], false, false);
        }
    }

    // Epilogue: D layout = (row = r + 8*(lane>>4), col = lane&15) per VGPR r.
    const int b  = mbase / TT;
    const int t0 = (mbase & (TT - 1)) + w * 16;
#pragma unroll
    for (int nt = 0; nt < 12; ++nt)
#pragma unroll
        for (int r = 0; r < 8; ++r) {
            const int t   = t0 + r + ((lane >> 4) << 3);
            const int col = ((nt & 3) << 4) + (lane & 15);
            const ushort_t hv = f2bf(acc[nt][r]);
            if (nt < 4) {
                qb[((size_t)b * TT + t) * HDIM + col] = hv;
            } else if (nt < 8) {
                kb[((size_t)b * TT + t) * HDIM + col] = hv;
            } else {
                vt[((size_t)b * HDIM + col) * TT + t] = hv;   // V transposed
            }
        }
}

// -----------------------------------------------------------------------------
// Kernel 2: causal flash attention per batch, async-LDS staged.
// Block = 256 threads = 8 waves, one block per batch.
// Key chunks of 32 are double-buffered into LDS with
// global_load_async_to_lds_b128 (2 loads/thread/chunk, s_wait_asynccnt 2).
// Wave w owns queries [32w, 32w+32): both row-tiles need chunks 0..w, so the
// lockstep chunk loop guards compute with (kc <= w), masking only at kc == w.
// -----------------------------------------------------------------------------
__global__ __launch_bounds__(256, 1)
void attn_kernel(const ushort_t* __restrict__ qb,
                 const ushort_t* __restrict__ kb,
                 const ushort_t* __restrict__ vt,
                 float* __restrict__ out) {
    __shared__ ushort_t kbuf[2][32 * HDIM];   // K chunk [32 keys x 64 d]   (2 x 4 KB)
    __shared__ ushort_t vbuf[2][HDIM * 32];   // V chunk [64 d x 32 keys]   (2 x 4 KB)
    __shared__ ushort_t pbuf[8][16 * 32];     // per-wave P re-layout       (8 KB)

    const int tid  = threadIdx.x;
    const int lane = tid & 31;
    const int w    = tid >> 5;
    const int b    = blockIdx.x;

    const ushort_t* qbB = qb + (size_t)b * TT * HDIM;
    const ushort_t* kbB = kb + (size_t)b * TT * HDIM;
    const ushort_t* vtB = vt + (size_t)b * HDIM * TT;
    float*          oB  = out + (size_t)b * TT * HDIM;

    const int grp8 = (lane >> 4) << 3;      // 0 or 8 (row offset in D layout)
    const int col  = lane & 15;

    // ---- Q fragments for both row-tiles, both 32-wide d-chunks ----
    Frag16 qf[2][2];
#pragma unroll
    for (int rt = 0; rt < 2; ++rt) {
        const int q  = w * 32 + rt * 16 + (lane & 15);
        const int c0 = (lane < 16) ? 0 : 16;
        const uint4* p = (const uint4*)&qbB[(size_t)q * HDIM];
#pragma unroll
        for (int dc = 0; dc < 2; ++dc) {
            qf[rt][dc].q2[0] = p[dc * 4 + (c0 >> 3)];
            qf[rt][dc].q2[1] = p[dc * 4 + (c0 >> 3) + 1];
        }
    }

    v8f o[2][4];
    float mrow[2][8], lrow[2][8];
#pragma unroll
    for (int rt = 0; rt < 2; ++rt) {
#pragma unroll
        for (int nt = 0; nt < 4; ++nt)
#pragma unroll
            for (int i = 0; i < 8; ++i) o[rt][nt][i] = 0.0f;
#pragma unroll
        for (int r = 0; r < 8; ++r) { mrow[rt][r] = -3.0e38f; lrow[rt][r] = 0.0f; }
    }

    // ---- async staging of a 32-key chunk: K 4KB + V 4KB, 2 b128 per thread ----
    // K chunk kc is contiguous: kbB[kc*32*64 ..]
    // V rows d have stride T: vtB[d*T + kc*32 ..]
    const int vd = tid >> 2, vp = tid & 3;
#define STAGE(KC, BUF)                                                          \
    do {                                                                        \
        async_b128(lds_off(&kbuf[BUF][tid * 8]),                                \
                   kbB + (size_t)(KC) * 32 * HDIM + tid * 8);                   \
        async_b128(lds_off(&vbuf[BUF][vd * 32 + vp * 8]),                       \
                   vtB + (size_t)vd * TT + (KC) * 32 + vp * 8);                 \
    } while (0)

    STAGE(0, 0);

    for (int kc = 0; kc < 8; ++kc) {
        const int buf = kc & 1;
        if (kc < 7) {
            STAGE(kc + 1, buf ^ 1);
            asm volatile("s_wait_asynccnt 2" ::: "memory");   // chunk kc landed
        } else {
            asm volatile("s_wait_asynccnt 0" ::: "memory");
        }
        __syncthreads();

        if (kc <= w) {
            const bool domask = (kc == w);
#pragma unroll
            for (int rt = 0; rt < 2; ++rt) {
                const int qbase = w * 32 + rt * 16;

                // ---- S = Q * K^T, two 16-wide key tiles from LDS ----
                v8f s[2];
#pragma unroll
                for (int t2 = 0; t2 < 2; ++t2)
#pragma unroll
                    for (int i = 0; i < 8; ++i) s[t2][i] = 0.0f;
#pragma unroll
                for (int t2 = 0; t2 < 2; ++t2) {
#pragma unroll
                    for (int dc = 0; dc < 2; ++dc) {
                        Frag16 kf;
                        const uint4* p = (const uint4*)
                            &kbuf[buf][(t2 * 16 + col) * HDIM + dc * 32
                                       + ((lane >> 4) << 4)];
                        kf.q2[0] = p[0];
                        kf.q2[1] = p[1];
                        s[t2] = __builtin_amdgcn_wmma_f32_16x16x32_bf16(
                            false, qf[rt][dc].v, false, kf.v, (short)0,
                            s[t2], false, false);
                    }
                }

                // ---- scale, causal mask, online softmax ----
#pragma unroll
                for (int r = 0; r < 8; ++r) {
                    const int q = qbase + r + grp8;
                    float sv0 = s[0][r] * 0.125f;        // 1/sqrt(64)
                    float sv1 = s[1][r] * 0.125f;
                    if (domask) {
                        if (kc * 32 + col      > q) sv0 = -3.0e38f;
                        if (kc * 32 + 16 + col > q) sv1 = -3.0e38f;
                    }
                    float mx = fmaxf(sv0, sv1);
#pragma unroll
                    for (int off = 1; off < 16; off <<= 1)
                        mx = fmaxf(mx, __shfl_xor(mx, off, 32));
                    const float mn = fmaxf(mrow[rt][r], mx);
                    const float p0 = __expf(sv0 - mn);
                    const float p1 = __expf(sv1 - mn);
                    float rs = p0 + p1;
#pragma unroll
                    for (int off = 1; off < 16; off <<= 1)
                        rs += __shfl_xor(rs, off, 32);
                    const float corr = __expf(mrow[rt][r] - mn);
                    lrow[rt][r] = lrow[rt][r] * corr + rs;
                    mrow[rt][r] = mn;
#pragma unroll
                    for (int nt = 0; nt < 4; ++nt) o[rt][nt][r] *= corr;
                    const int row = r + grp8;
                    pbuf[w][row * 32 + col]      = f2bf(p0);
                    pbuf[w][row * 32 + 16 + col] = f2bf(p1);
                }

                // ---- reload P as 16x32 A fragment ----
                Frag16 pf;
                {
                    const int c0 = (lane < 16) ? 0 : 16;
                    const uint4* p = (const uint4*)&pbuf[w][(lane & 15) * 32];
                    pf.q2[0] = p[(c0 >> 3)];
                    pf.q2[1] = p[(c0 >> 3) + 1];
                }

                // ---- O += P * V from LDS ----
#pragma unroll
                for (int nt = 0; nt < 4; ++nt) {
                    Frag16 vf;
                    const uint4* p = (const uint4*)
                        &vbuf[buf][(nt * 16 + col) * 32 + ((lane >> 4) << 4)];
                    vf.q2[0] = p[0];
                    vf.q2[1] = p[1];
                    o[rt][nt] = __builtin_amdgcn_wmma_f32_16x16x32_bf16(
                        false, pf.v, false, vf.v, (short)0, o[rt][nt],
                        false, false);
                }
            }
        }
        __syncthreads();   // consumers done before next issue overwrites buf^1
    }
#undef STAGE

    // ---- epilogue: normalize by l and store fp32 ----
#pragma unroll
    for (int rt = 0; rt < 2; ++rt)
#pragma unroll
        for (int nt = 0; nt < 4; ++nt)
#pragma unroll
            for (int r = 0; r < 8; ++r) {
                const int q = w * 32 + rt * 16 + r + grp8;
                oB[(size_t)q * HDIM + nt * 16 + col] = o[rt][nt][r] / lrow[rt][r];
            }
}

// -----------------------------------------------------------------------------
// Host launcher
// -----------------------------------------------------------------------------
extern "C" void kernel_launch(void* const* d_in, const int* in_sizes, int n_in,
                              void* d_out, int out_size, void* d_ws, size_t ws_size,
                              hipStream_t stream) {
    const float* x  = (const float*)d_in[0];   // [B,T,384]
    const float* wq = (const float*)d_in[1];   // [64,384]
    const float* wk = (const float*)d_in[2];
    const float* wv = (const float*)d_in[3];
    float* out = (float*)d_out;                // [B,T,64] fp32

    const size_t elems = (size_t)BB * TT * HDIM;   // 8,388,608
    ushort_t* qb = (ushort_t*)d_ws;
    ushort_t* kb = qb + elems;
    ushort_t* vt = kb + elems;

    qkv_proj_kernel<<<(BB * TT) / 128, 256, 0, stream>>>(x, wq, wk, wv, qb, kb, vt);
    attn_kernel<<<BB, 256, 0, stream>>>(qb, kb, vt, out);
}
